// matcher_41918880809174
// MI455X (gfx1250) — matrix operand order repaired and checked by reading, not compiled, not run on hardware
//
#include <hip/hip_runtime.h>
#include <cstddef>

typedef __bf16 bf16_t;
typedef bf16_t v16bf __attribute__((ext_vector_type(16)));
typedef bf16_t v8bf  __attribute__((ext_vector_type(8)));
typedef bf16_t bf16x2 __attribute__((ext_vector_type(2)));
typedef float  v8f   __attribute__((ext_vector_type(8)));
typedef unsigned u32x4 __attribute__((ext_vector_type(4)));

#define NPAD 2048
#define NPTS 2000
#define NB   2
#define NC   128
#define NHEAD 4
#define NS   2001

#define GM_PLAIN 0
#define GM_BNRELU 1
#define GM_ADD 2

__device__ inline v16bf cat8(u32x4 a, u32x4 b) {
  v8bf x = __builtin_bit_cast(v8bf, a), y = __builtin_bit_cast(v8bf, b);
  v16bf r;
  #pragma unroll
  for (int i = 0; i < 8; ++i) { r[i] = x[i]; r[8 + i] = y[i]; }
  return r;
}

// ---------------- normalize + transpose: desc [B,N,C] -> [B,C,NPAD], pad=0 ----
__global__ __launch_bounds__(128)
void k_norm_t(const float* __restrict__ desc, float* __restrict__ out) {
  int n = blockIdx.x, b = blockIdx.y, c = threadIdx.x;
  float v = (n < NPTS) ? desc[((size_t)b * NPTS + n) * NC + c] : 0.f;
  float ss = v * v;
  #pragma unroll
  for (int m = 1; m < 32; m <<= 1) ss += __shfl_xor(ss, m, 32);
  __shared__ float red[4];
  if ((threadIdx.x & 31) == 0) red[threadIdx.x >> 5] = ss;
  __syncthreads();
  float tot = red[0] + red[1] + red[2] + red[3];
  float nrm = fmaxf(sqrtf(tot), 1e-12f);
  out[((size_t)b * NC + c) * NPAD + n] = v / nrm;
}

// ---------------- x [B,N,2] -> e [B,2,NPAD], pad=0 --------------------------
__global__ void k_xpose(const float* __restrict__ x, float* __restrict__ e) {
  int n = blockIdx.x * blockDim.x + threadIdx.x, b = blockIdx.y;
  if (n >= NPAD) return;
  #pragma unroll
  for (int c = 0; c < 2; ++c)
    e[((size_t)b * 2 + c) * NPAD + n] = (n < NPTS) ? x[((size_t)b * NPTS + n) * 2 + c] : 0.f;
}

// ---------------- small scalar conv + bn + relu (cin=2) ---------------------
__global__ __launch_bounds__(64)
void k_conv_small(const float* __restrict__ x, const float* __restrict__ w,
                  const float* __restrict__ bias, const float* __restrict__ bnb,
                  const float* __restrict__ bng, const float* __restrict__ bnm,
                  const float* __restrict__ bnv, float* __restrict__ y,
                  int cin, int cout) {
  int n = blockIdx.x * 64 + threadIdx.x;
  int co = blockIdx.y, b = blockIdx.z;
  float acc = bias[co];
  for (int ci = 0; ci < cin; ++ci)
    acc += w[co * cin + ci] * x[((size_t)b * cin + ci) * NPAD + n];
  float s = bng[co] * rsqrtf(bnv[co] + 1e-5f);
  acc = (acc - bnm[co]) * s + bnb[co];
  y[((size_t)b * cout + co) * NPAD + n] = fmaxf(acc, 0.f);
}

// ---------------- WMMA bf16 1x1-conv GEMM: Y = W@X (+bias, bn/relu/add) -----
// wave computes 16(co) x 64(n); block = 4 waves = 64(co) x 64(n)
// Output modes: fp32 (Y), interleaved bf16 (Ybf), head-transposed bf16 (Ytr:
// layout [B][H][NPAD][32], co = d*NHEAD+h -> [b][co&3][n][co>>2]).
__global__ __launch_bounds__(128)
void k_gemm(const float* __restrict__ W, const float* __restrict__ bias,
            const float* __restrict__ X, float* __restrict__ Y,
            bf16_t* __restrict__ Ybf, bf16_t* __restrict__ Ytr,
            const float* __restrict__ bnb, const float* __restrict__ bng,
            const float* __restrict__ bnm, const float* __restrict__ bnv,
            const float* __restrict__ addend, int Cin, int Cout, int mode) {
  int lane = threadIdx.x & 31, wave = threadIdx.x >> 5;
  int b = blockIdx.z;
  int co0 = blockIdx.y * 64 + wave * 16;
  int n0 = blockIdx.x * 64;
  int rl = lane & 15, hi = lane >> 4;
  const float* Xb = X + (size_t)b * Cin * NPAD;
  v8f acc[4] = {};
  for (int kk = 0; kk < Cin; kk += 32) {
    v16bf a;
    const float* wrow = W + (size_t)(co0 + rl) * Cin + kk;
    #pragma unroll
    for (int i = 0; i < 8; ++i) {
      int k = hi * 16 + 2 * i;
      float2 wp = *(const float2*)(wrow + k);
      a[2 * i] = (bf16_t)wp.x; a[2 * i + 1] = (bf16_t)wp.y;
    }
    #pragma unroll
    for (int t = 0; t < 4; ++t) {
      int col = n0 + t * 16 + rl;
      v16bf bb;
      #pragma unroll
      for (int j = 0; j < 8; ++j) {
        int k = kk + hi * 16 + 2 * j;
        bb[2 * j]     = (bf16_t)Xb[(size_t)k * NPAD + col];
        bb[2 * j + 1] = (bf16_t)Xb[(size_t)(k + 1) * NPAD + col];
      }
      acc[t] = __builtin_amdgcn_wmma_f32_16x16x32_bf16(false, a, false, bb,
                                                       (short)0, acc[t], false, false);
    }
  }
  #pragma unroll
  for (int t = 0; t < 4; ++t) {
    int col = n0 + t * 16 + rl;
    #pragma unroll
    for (int r = 0; r < 8; ++r) {
      int co = co0 + r + 8 * hi;
      float val = acc[t][r] + bias[co];
      if (mode == GM_BNRELU) {
        float s = bng[co] * rsqrtf(bnv[co] + 1e-5f);
        val = fmaxf((val - bnm[co]) * s + bnb[co], 0.f);
      } else if (mode == GM_ADD) {
        val += addend[(size_t)b * Cout * NPAD + (size_t)co * NPAD + col];
      }
      if (Ytr) {
        Ytr[(((size_t)b * NHEAD + (co & 3)) * NPAD + col) * 32 + (co >> 2)] = (bf16_t)val;
      } else {
        size_t oidx = (size_t)b * Cout * NPAD + (size_t)co * NPAD + col;
        if (Ybf) Ybf[oidx] = (bf16_t)val;
        else     Y[oidx] = val;
      }
    }
  }
}

// ---------------- flash attention per (b, head, 16-query tile) --------------
// QT/KT: head-transposed bf16 [B][H][NPAD][32]; V: interleaved bf16
// [B][C][NPAD] with c = d*NHEAD+h.  Computes S^T = K_h^T Q_h tiles, online
// softmax over keys, O^T = V_h P^T.  All fragment loads are b128.
__global__ __launch_bounds__(128)
void k_flash(const bf16_t* __restrict__ QT, const bf16_t* __restrict__ KT,
             const bf16_t* __restrict__ V, float* __restrict__ O, int Mkeys) {
  int lane = threadIdx.x & 31, wave = threadIdx.x >> 5;
  int h = blockIdx.y, b = blockIdx.z;
  int n0 = blockIdx.x * 64 + wave * 16;
  int col = lane & 15, hi = lane >> 4;
  const bf16_t* QTb = QT + ((size_t)(b * NHEAD + h)) * NPAD * 32;
  const bf16_t* KTb = KT + ((size_t)(b * NHEAD + h)) * NPAD * 32;
  const bf16_t* Vb  = V + (size_t)b * NC * NPAD;
  const float scale = 0.1767766953f; // 1/sqrt(32)

  // Q B-frag: dwords 0-3 = d hi*16..+7, dwords 4-7 = d hi*16+8..+15
  const bf16_t* qrow = QTb + (size_t)(n0 + col) * 32 + hi * 16;
  v16bf bq = cat8(*(const u32x4*)qrow, *(const u32x4*)(qrow + 8));

  v8f o0 = {}, o1 = {};
  float run_max = -1e30f, run_sum = 0.f;

  for (int mb = 0; mb < Mkeys; mb += 32) {
    if (mb + 32 < Mkeys) {
      __builtin_prefetch(KTb + (size_t)(mb + 32 + lane) * 32, 0, 0);
      __builtin_prefetch(Vb + ((size_t)(lane * NHEAD + h)) * NPAD + mb + 32, 0, 0);
    }
    float p[2][8];
    float cmax = -1e30f;
    #pragma unroll
    for (int t = 0; t < 2; ++t) {
      // K A-frag: dwords 0-3 = d hi*16..+7, dwords 4-7 = d hi*16+8..+15
      const bf16_t* krow = KTb + (size_t)(mb + t * 16 + col) * 32 + hi * 16;
      v16bf ak = cat8(*(const u32x4*)krow, *(const u32x4*)(krow + 8));
      v8f z = {};
      v8f st = __builtin_amdgcn_wmma_f32_16x16x32_bf16(false, ak, false, bq,
                                                       (short)0, z, false, false);
      #pragma unroll
      for (int r = 0; r < 8; ++r) {
        int m2 = mb + t * 16 + r + 8 * hi;
        float s = (m2 < Mkeys) ? st[r] * scale : -1e30f;
        p[t][r] = s;
        cmax = fmaxf(cmax, s);
      }
    }
    cmax = fmaxf(cmax, __shfl_xor(cmax, 16, 32));
    float nmax = fmaxf(run_max, cmax);
    float alpha = __expf(run_max - nmax);
    float csum = 0.f;
    #pragma unroll
    for (int t = 0; t < 2; ++t)
      #pragma unroll
      for (int r = 0; r < 8; ++r) { float e = __expf(p[t][r] - nmax); p[t][r] = e; csum += e; }
    csum += __shfl_xor(csum, 16, 32);
    run_sum = run_sum * alpha + csum;
    run_max = nmax;
    #pragma unroll
    for (int r = 0; r < 8; ++r) { o0[r] *= alpha; o1[r] *= alpha; }

    // pack probabilities to bf16x2, then shuffle packed words into B-frag
    float pk0[4], pk1[4];
    #pragma unroll
    for (int q = 0; q < 4; ++q) {
      bf16x2 t0; t0[0] = (bf16_t)p[0][2 * q]; t0[1] = (bf16_t)p[0][2 * q + 1];
      bf16x2 t1; t1[0] = (bf16_t)p[1][2 * q]; t1[1] = (bf16_t)p[1][2 * q + 1];
      pk0[q] = __builtin_bit_cast(float, t0);
      pk1[q] = __builtin_bit_cast(float, t1);
    }
    union { v16bf v; float f[8]; } bpu;
    #pragma unroll
    for (int j = 0; j < 8; ++j) {
      int src = (j >> 2) * 16 + col;
      float u0 = __shfl(pk0[j & 3], src, 32);
      float u1 = __shfl(pk1[j & 3], src, 32);
      bpu.f[j] = hi ? u1 : u0;
    }
    v16bf bp = bpu.v;

    // O^T += V_h * P^T : two A-frags (d rows 0-15, 16-31); m runs contiguous
    #pragma unroll
    for (int half = 0; half < 2; ++half) {
      int d = half * 16 + col;
      const bf16_t* vrow = Vb + ((size_t)(d * NHEAD + h)) * NPAD + mb + hi * 16;
      v16bf av = cat8(*(const u32x4*)vrow, *(const u32x4*)(vrow + 8));
      if (half == 0)
        o0 = __builtin_amdgcn_wmma_f32_16x16x32_bf16(false, av, false, bp, (short)0, o0, false, false);
      else
        o1 = __builtin_amdgcn_wmma_f32_16x16x32_bf16(false, av, false, bp, (short)0, o1, false, false);
    }
  }
  float inv = 1.0f / run_sum;
  float* Ob = O + (size_t)b * NC * NPAD;
  #pragma unroll
  for (int r = 0; r < 8; ++r) {
    int d = r + 8 * hi;
    Ob[((size_t)(d * NHEAD + h)) * NPAD + n0 + col] = o0[r] * inv;
    int d2 = 16 + r + 8 * hi;
    Ob[((size_t)(d2 * NHEAD + h)) * NPAD + n0 + col] = o1[r] * inv;
  }
}

// ---------------- concat [fea ; mh] along channels --------------------------
__global__ void k_concat(const float* __restrict__ A, const float* __restrict__ Bm,
                         float* __restrict__ Out) {
  size_t idx = (size_t)blockIdx.x * blockDim.x + threadIdx.x;
  size_t per = (size_t)NC * NPAD;
  size_t b = idx / per, rem = idx % per;
  Out[b * 2 * per + rem] = A[idx];
  Out[b * 2 * per + per + rem] = Bm[idx];
}

// ---------------- desc_mat = a1^T a2 (WMMA, K=128, bf16 inputs) -------------
__global__ __launch_bounds__(128)
void k_descmat(const bf16_t* __restrict__ A1, const bf16_t* __restrict__ A2,
               float* __restrict__ out) {
  int lane = threadIdx.x & 31, wave = threadIdx.x >> 5;
  int b = blockIdx.z;
  int n0 = blockIdx.y * 64 + wave * 16;
  int m0 = blockIdx.x * 64;
  int rl = lane & 15, hi = lane >> 4;
  const bf16_t* X1 = A1 + (size_t)b * NC * NPAD;
  const bf16_t* X2 = A2 + (size_t)b * NC * NPAD;
  v8f acc[4] = {};
  for (int kk = 0; kk < NC; kk += 32) {
    v16bf a;
    int nrow = n0 + rl;
    #pragma unroll
    for (int i = 0; i < 8; ++i) {
      int k = kk + hi * 16 + 2 * i;
      a[2 * i]     = X1[(size_t)k * NPAD + nrow];
      a[2 * i + 1] = X1[(size_t)(k + 1) * NPAD + nrow];
    }
    #pragma unroll
    for (int t = 0; t < 4; ++t) {
      int mcol = m0 + t * 16 + rl;
      v16bf bb;
      #pragma unroll
      for (int j = 0; j < 8; ++j) {
        int k = kk + hi * 16 + 2 * j;
        bb[2 * j]     = X2[(size_t)k * NPAD + mcol];
        bb[2 * j + 1] = X2[(size_t)(k + 1) * NPAD + mcol];
      }
      acc[t] = __builtin_amdgcn_wmma_f32_16x16x32_bf16(false, a, false, bb,
                                                       (short)0, acc[t], false, false);
    }
  }
  #pragma unroll
  for (int t = 0; t < 4; ++t) {
    int m = m0 + t * 16 + rl;
    #pragma unroll
    for (int r = 0; r < 8; ++r) {
      int n = n0 + r + 8 * hi;
      if (n < NPTS && m < NPTS)
        out[((size_t)b * NS + n) * NS + m] = acc[t][r];
    }
  }
}

// ---------------- dustbin fill ----------------------------------------------
__global__ void k_dustbin(float* __restrict__ out, const float* __restrict__ dust) {
  float d = dust[0];
  int i = blockIdx.x * 256 + threadIdx.x, b = blockIdx.y;
  float* p = out + (size_t)b * NS * NS;
  if (i < NPTS) p[(size_t)i * NS + NPTS] = d;
  if (i < NS)   p[(size_t)NPTS * NS + i] = d;
}

// ---------------- in-place row softmax (width 2001) -------------------------
__global__ __launch_bounds__(256)
void k_rowsoftmax(float* __restrict__ out) {
  int n = blockIdx.x, b = blockIdx.y;
  float* row = out + ((size_t)b * NS + n) * NS;
  __shared__ float red[8];
  int tid = threadIdx.x, lane = tid & 31, w = tid >> 5;
  float mx = -1e30f;
  for (int m = tid; m < NS; m += 256) mx = fmaxf(mx, row[m]);
  #pragma unroll
  for (int s = 16; s > 0; s >>= 1) mx = fmaxf(mx, __shfl_xor(mx, s, 32));
  if (!lane) red[w] = mx;
  __syncthreads();
  mx = red[0];
  #pragma unroll
  for (int i = 1; i < 8; ++i) mx = fmaxf(mx, red[i]);
  __syncthreads();
  float sum = 0.f;
  for (int m = tid; m < NS; m += 256) { float e = __expf(row[m] - mx); row[m] = e; sum += e; }
  #pragma unroll
  for (int s = 16; s > 0; s >>= 1) sum += __shfl_xor(sum, s, 32);
  if (!lane) red[w] = sum;
  __syncthreads();
  float tot = 0.f;
  #pragma unroll
  for (int i = 0; i < 8; ++i) tot += red[i];
  float inv = 1.0f / tot;
  for (int m = tid; m < NS; m += 256) row[m] *= inv;
}

// ---------------- sinkhorn --------------------------------------------------
__global__ void k_init_uv(float* __restrict__ u, float* __restrict__ v) {
  int i = blockIdx.x * 256 + threadIdx.x;
  if (i < NB * NS) { u[i] = 1.f; v[i] = 1.f; }
}

__global__ __launch_bounds__(256)
void k_sink_u(const float* __restrict__ p, const float* __restrict__ v,
              float* __restrict__ u) {
  int n = blockIdx.x, b = blockIdx.y;
  const float* row = p + ((size_t)b * NS + n) * NS;
  const float* vb = v + b * NS;
  __shared__ float red[8];
  int tid = threadIdx.x, lane = tid & 31, w = tid >> 5;
  float acc = 0.f;
  for (int m = tid; m < NS; m += 256) acc += row[m] * vb[m];
  #pragma unroll
  for (int s = 16; s > 0; s >>= 1) acc += __shfl_xor(acc, s, 32);
  if (!lane) red[w] = acc;
  __syncthreads();
  if (tid == 0) {
    float tot = 0.f;
    #pragma unroll
    for (int i = 0; i < 8; ++i) tot += red[i];
    float r = (n < NPTS) ? 1.f : (float)(NPTS + 1);
    u[b * NS + n] = r / (tot + 1e-8f);
  }
}

__global__ __launch_bounds__(256)
void k_sink_v(const float* __restrict__ p, const float* __restrict__ u,
              float* __restrict__ v) {
  int m = blockIdx.x * 256 + threadIdx.x, b = blockIdx.y;
  if (m >= NS) return;
  const float* pb = p + (size_t)b * NS * NS;
  const float* ub = u + b * NS;
  float acc = 0.f;
  #pragma unroll 4
  for (int n = 0; n < NS; ++n) acc += pb[(size_t)n * NS + m] * ub[n];
  float c = (m < NPTS) ? 1.f : (float)(NPTS + 1);
  v[b * NS + m] = c / (acc + 1e-8f);
}

__global__ void k_scale(float* __restrict__ out, const float* __restrict__ u,
                        const float* __restrict__ v) {
  size_t idx = (size_t)blockIdx.x * 256 + threadIdx.x;
  size_t per = (size_t)NS * NS;
  if (idx >= per) return;
  int b = blockIdx.y;
  int n = (int)(idx / NS), m = (int)(idx % NS);
  out[(size_t)b * per + idx] *= u[b * NS + n] * v[b * NS + m];
}

// ============================ host orchestration ============================
extern "C" void kernel_launch(void* const* d_in, const int* in_sizes, int n_in,
                              void* d_out, int out_size, void* d_ws, size_t ws_size,
                              hipStream_t stream) {
  (void)in_sizes; (void)n_in; (void)out_size; (void)ws_size;
  auto F = [&](int i) { return (const float*)d_in[i]; };

  // pytree leaf order (sorted dict keys): desc1, desc2, x1, x2,
  //   params{cross[9x16], final(b,w), pos(26), self[9x16]}, dustbin
  const int IDX_DESC1 = 0, IDX_DESC2 = 1, IDX_X1 = 2, IDX_X2 = 3;
  const int PBASE = 4;
  const int CROSS0 = PBASE;
  const int FINAL_B = PBASE + 144, FINAL_W = PBASE + 145;
  const int POS0 = PBASE + 146;
  const int SELF0 = PBASE + 172;
  const int IDX_DUST = PBASE + 316;
  // per-attn leaf offsets (sorted: af1,af2,af_bn,k,mh,q,v; conv:(b,w); bn:(beta,g,m,v))
  const int AF1_B=0, AF1_W=1, AF2_B=2, AF2_W=3, BN_BETA=4, BN_G=5, BN_M=6, BN_V=7,
            K_B=8, K_W=9, MH_B=10, MH_W=11, Q_B=12, Q_W=13, V_B=14, V_W=15;

  float* Wks = (float*)d_ws;
  const size_t SZ1 = (size_t)NB * NC * NPAD; // 524288 floats
  float* A1  = Wks;            float* A2  = Wks + SZ1;
  // bf16 buffers overlay float-sized slots (half used)
  bf16_t* QT1 = (bf16_t*)(Wks + 2 * SZ1);  // head-transposed [B][H][NPAD][32]
  bf16_t* KT1 = (bf16_t*)(Wks + 3 * SZ1);
  bf16_t* VB1 = (bf16_t*)(Wks + 4 * SZ1);  // interleaved [B][C][NPAD]
  bf16_t* QT2 = (bf16_t*)(Wks + 5 * SZ1);
  bf16_t* KT2 = (bf16_t*)(Wks + 6 * SZ1);
  bf16_t* VB2 = (bf16_t*)(Wks + 7 * SZ1);
  float* ATT1= Wks + 8 * SZ1;  float* ATT2= Wks + 9 * SZ1;
  float* MH1 = Wks + 10 * SZ1; float* MH2 = Wks + 11 * SZ1;
  float* CAT1= Wks + 12 * SZ1; float* CAT2= Wks + 14 * SZ1;
  float* HID1= Wks + 16 * SZ1; float* HID2= Wks + 18 * SZ1;
  float* E1  = Wks + 20 * SZ1; float* E2  = E1 + (size_t)NB * 2 * NPAD;
  float* Uv  = E2 + (size_t)NB * 2 * NPAD; float* Vv = Uv + 4096;
  float* OUT = (float*)d_out;

  auto gemm = [&](const float* W, const float* bias, const float* X, float* Y,
                  bf16_t* Ybf, bf16_t* Ytr, int Cin, int Cout, int mode,
                  const float* bnb, const float* bng, const float* bnm,
                  const float* bnv, const float* add) {
    k_gemm<<<dim3(NPAD / 64, Cout / 64, NB), 128, 0, stream>>>(
        W, bias, X, Y, Ybf, Ytr, bnb, bng, bnm, bnv, add, Cin, Cout, mode);
  };

  // 1) normalize + transpose descriptors
  k_norm_t<<<dim3(NPAD, NB), 128, 0, stream>>>(F(IDX_DESC1), A1);
  k_norm_t<<<dim3(NPAD, NB), 128, 0, stream>>>(F(IDX_DESC2), A2);
  k_xpose<<<dim3(NPAD / 256, NB), 256, 0, stream>>>(F(IDX_X1), E1);
  k_xpose<<<dim3(NPAD / 256, NB), 256, 0, stream>>>(F(IDX_X2), E2);

  // 2) positional encoder (conv0 scalar, conv1..4 WMMA), += normalized desc
  auto pos_enc = [&](const float* E, float* ping, float* pong, float* Aout) {
    k_conv_small<<<dim3(NPAD / 64, 32, NB), 64, 0, stream>>>(
        E, F(POS0 + 1), F(POS0 + 0), F(POS0 + 2), F(POS0 + 3), F(POS0 + 4), F(POS0 + 5),
        ping, 2, 32);
    gemm(F(POS0 + 7),  F(POS0 + 6),  ping, pong, nullptr, nullptr, 32, 64, GM_BNRELU,
         F(POS0 + 8),  F(POS0 + 9),  F(POS0 + 10), F(POS0 + 11), nullptr);
    gemm(F(POS0 + 13), F(POS0 + 12), pong, ping, nullptr, nullptr, 64, 128, GM_BNRELU,
         F(POS0 + 14), F(POS0 + 15), F(POS0 + 16), F(POS0 + 17), nullptr);
    gemm(F(POS0 + 19), F(POS0 + 18), ping, pong, nullptr, nullptr, 128, 256, GM_BNRELU,
         F(POS0 + 20), F(POS0 + 21), F(POS0 + 22), F(POS0 + 23), nullptr);
    gemm(F(POS0 + 25), F(POS0 + 24), pong, Aout, nullptr, nullptr, 256, 128, GM_ADD,
         nullptr, nullptr, nullptr, nullptr, Aout);
  };
  pos_enc(E1, CAT1, HID1, A1);
  pos_enc(E2, CAT2, HID2, A2);

  // 3) attention layers
  auto attn = [&](int ab, bool cross) {
    gemm(F(ab + Q_W), F(ab + Q_B), A1, nullptr, nullptr, QT1, NC, NC, GM_PLAIN, nullptr, nullptr, nullptr, nullptr, nullptr);
    gemm(F(ab + K_W), F(ab + K_B), A1, nullptr, nullptr, KT1, NC, NC, GM_PLAIN, nullptr, nullptr, nullptr, nullptr, nullptr);
    gemm(F(ab + V_W), F(ab + V_B), A1, nullptr, VB1, nullptr, NC, NC, GM_PLAIN, nullptr, nullptr, nullptr, nullptr, nullptr);
    gemm(F(ab + Q_W), F(ab + Q_B), A2, nullptr, nullptr, QT2, NC, NC, GM_PLAIN, nullptr, nullptr, nullptr, nullptr, nullptr);
    gemm(F(ab + K_W), F(ab + K_B), A2, nullptr, nullptr, KT2, NC, NC, GM_PLAIN, nullptr, nullptr, nullptr, nullptr, nullptr);
    gemm(F(ab + V_W), F(ab + V_B), A2, nullptr, VB2, nullptr, NC, NC, GM_PLAIN, nullptr, nullptr, nullptr, nullptr, nullptr);
    dim3 fg(NPAD / 64, NHEAD, NB);
    if (!cross) {
      k_flash<<<fg, 128, 0, stream>>>(QT1, KT1, VB1, ATT1, NPTS);
      k_flash<<<fg, 128, 0, stream>>>(QT2, KT2, VB2, ATT2, NPTS);
    } else {
      k_flash<<<fg, 128, 0, stream>>>(QT1, KT2, VB2, ATT1, NPTS);
      k_flash<<<fg, 128, 0, stream>>>(QT2, KT1, VB1, ATT2, NPTS);
    }
    gemm(F(ab + MH_W), F(ab + MH_B), ATT1, MH1, nullptr, nullptr, NC, NC, GM_PLAIN, nullptr, nullptr, nullptr, nullptr, nullptr);
    gemm(F(ab + MH_W), F(ab + MH_B), ATT2, MH2, nullptr, nullptr, NC, NC, GM_PLAIN, nullptr, nullptr, nullptr, nullptr, nullptr);
    k_concat<<<dim3((unsigned)(SZ1 / 256)), 256, 0, stream>>>(A1, MH1, CAT1);
    k_concat<<<dim3((unsigned)(SZ1 / 256)), 256, 0, stream>>>(A2, MH2, CAT2);
    gemm(F(ab + AF1_W), F(ab + AF1_B), CAT1, HID1, nullptr, nullptr, 256, 256, GM_BNRELU,
         F(ab + BN_BETA), F(ab + BN_G), F(ab + BN_M), F(ab + BN_V), nullptr);
    gemm(F(ab + AF1_W), F(ab + AF1_B), CAT2, HID2, nullptr, nullptr, 256, 256, GM_BNRELU,
         F(ab + BN_BETA), F(ab + BN_G), F(ab + BN_M), F(ab + BN_V), nullptr);
    gemm(F(ab + AF2_W), F(ab + AF2_B), HID1, A1, nullptr, nullptr, 256, 128, GM_ADD,
         nullptr, nullptr, nullptr, nullptr, A1);
    gemm(F(ab + AF2_W), F(ab + AF2_B), HID2, A2, nullptr, nullptr, 256, 128, GM_ADD,
         nullptr, nullptr, nullptr, nullptr, A2);
  };
  for (int l = 0; l < 9; ++l) {
    attn(SELF0 + l * 16, false);
    attn(CROSS0 + l * 16, true);
  }

  // 4) final conv (interleaved bf16 out -> feeds score-matrix WMMA)
  bf16_t* FA1 = QT1; bf16_t* FA2 = QT2;
  gemm(F(FINAL_W), F(FINAL_B), A1, nullptr, FA1, nullptr, NC, NC, GM_PLAIN, nullptr, nullptr, nullptr, nullptr, nullptr);
  gemm(F(FINAL_W), F(FINAL_B), A2, nullptr, FA2, nullptr, NC, NC, GM_PLAIN, nullptr, nullptr, nullptr, nullptr, nullptr);

  // 5) score matrix + dustbin + softmax + sinkhorn (in-place in d_out)
  k_descmat<<<dim3(NPAD / 64, NPAD / 64, NB), 128, 0, stream>>>(FA1, FA2, OUT);
  k_dustbin<<<dim3((NS + 255) / 256, NB), 256, 0, stream>>>(OUT, F(IDX_DUST));
  k_rowsoftmax<<<dim3(NS, NB), 256, 0, stream>>>(OUT);
  k_init_uv<<<dim3((NB * NS + 255) / 256), 256, 0, stream>>>(Uv, Vv);
  for (int it = 0; it < 100; ++it) {
    k_sink_u<<<dim3(NS, NB), 256, 0, stream>>>(OUT, Vv, Uv);
    k_sink_v<<<dim3((NS + 255) / 256, NB), 256, 0, stream>>>(OUT, Uv, Vv);
  }
  k_scale<<<dim3((unsigned)(((size_t)NS * NS + 255) / 256), NB), 256, 0, stream>>>(OUT, Uv, Vv);
}
